// HGTEncoder_12257836662991
// MI455X (gfx1250) — matrix operand order repaired and checked by reading, not compiled, not run on hardware
//
#include <hip/hip_runtime.h>

// ---------------------------------------------------------------------------
// HGT encoder for MI455X (gfx1250, wave32, WMMA).
// Dense projections via v_wmma_f32_16x16x32_bf16 (bf16 in, f32 acc),
// double-buffered LDS staging, packed bf16x2 DS stores.
// Edge message transform also via WMMA; segment softmax via f32 atomics.
// All hot-path addressing uses uniform base + 32-bit offsets (SADDR+VGPR form).
// ---------------------------------------------------------------------------

#define TT 3          // node types
#define NR 6          // relations
#define NN 100000     // nodes per type
#define NE 150000     // edges per relation (divisible by 16)
#define ND 128        // hidden dim
#define NH 4          // heads
#define NDK 32        // dim per head
#define NL 2          // layers

typedef __attribute__((ext_vector_type(16))) __bf16          v16bf;
typedef __attribute__((ext_vector_type(8)))  float           v8f;
typedef __attribute__((ext_vector_type(8)))  unsigned short  us8;
typedef __attribute__((ext_vector_type(16))) unsigned short  us16;

__device__ __forceinline__ unsigned short f2bf(float f) {
    unsigned u = __float_as_uint(f);
    unsigned r = u + 0x7FFFu + ((u >> 16) & 1u);   // round-to-nearest-even
    return (unsigned short)(r >> 16);
}
__device__ __forceinline__ unsigned pk2bf(float a, float b) {
    return (unsigned)f2bf(a) | ((unsigned)f2bf(b) << 16);
}

__device__ __forceinline__ v16bf mkfrag(us8 lo, us8 hi) {
    us16 t = __builtin_shufflevector(lo, hi, 0,1,2,3,4,5,6,7,8,9,10,11,12,13,14,15);
    return __builtin_bit_cast(v16bf, t);
}

__device__ __forceinline__ void atomicMaxF32(float* addr, float val) {
    unsigned* ua = (unsigned*)addr;
    unsigned old = *ua;
    while (true) {
        float f = __uint_as_float(old);
        if (f >= val) break;
        unsigned assumed = old;
        old = atomicCAS(ua, assumed, __float_as_uint(val));
        if (old == assumed) break;
    }
}

// ---------------------------------------------------------------------------
// Generic WMMA GEMM:  Out[n, 0:128] = epilogue( (scale_n * A[g(n), :]) @ W + b )
// Block: 256 threads (8 waves), tile 128 rows x 128 cols, K in 4 steps of 32,
// double-buffered LDS so global loads of step k+1 overlap WMMAs of step k.
// MODE 0: +bias   MODE 1: tanh(+bias)
// MODE 2: (+bias)*sigmoid(skip) + Xprev*(1-sigmoid(skip))   (A-proj + residual)
// ---------------------------------------------------------------------------
#define LDS_STR 40   // ushort stride (80B = 20 banks: conflict-free frag reads)

template<int MODE, bool GATHER, bool SCALE>
__global__ __launch_bounds__(256)
void wmma_gemm_kernel(const float* __restrict__ A,
                      const int*   __restrict__ gidx,      // row gather (GATHER)
                      const float* __restrict__ rowScale,  // 1/max(cnt,1) (SCALE)
                      const float* __restrict__ W,         // [128][128] (K x N)
                      const float* __restrict__ bias,      // [128]
                      const float* __restrict__ Xprev,     // MODE 2 residual
                      const float* __restrict__ skipPtr,   // MODE 2 skip scalar
                      float*       __restrict__ Out,
                      int nrows)
{
    __shared__ __align__(16) unsigned short lA[2][128 * LDS_STR];
    __shared__ __align__(16) unsigned short lB[2][128 * LDS_STR]; // col-major [n][k]

    const int tid  = threadIdx.x;
    const int wv   = tid >> 5;
    const int lane = tid & 31;
    const int hi16 = lane >> 4;      // 0 | 1
    const int l16  = lane & 15;
    const int rowBase = blockIdx.x * 128;

    // staging geometry (fixed per thread)
    const int rA = tid >> 3;          // A row slot (0..31, +32/pass)
    const int cA = (tid & 7) * 4;     // A k columns (float4)
    const int kB = tid >> 5;          // B k slot (0..7, +8/pass)
    const int nB = (tid & 31) * 4;    // B n columns (float4)

    // per-pass row metadata (independent of k-step); 32-bit element offsets
    unsigned aoff[4];
    bool avalid[4];
    float scv[4];
    #pragma unroll
    for (int p = 0; p < 4; ++p) {
        const int grow = rowBase + rA + 32 * p;
        avalid[p] = (grow < nrows);
        unsigned arow = 0;
        float sc = 1.0f;
        if (avalid[p]) {
            arow = GATHER ? (unsigned)gidx[grow] : (unsigned)grow;
            if (SCALE) sc = 1.0f / fmaxf(rowScale[grow], 1.0f);
        }
        aoff[p] = arow * ND + cA;
        scv[p]  = sc;
    }

    float aReg[4][4], bReg[4][4];

    auto loadGlobal = [&](int ks) {
        #pragma unroll
        for (int p = 0; p < 4; ++p) {
            float4 v = make_float4(0.f, 0.f, 0.f, 0.f);
            if (avalid[p]) v = *(const float4*)(A + aoff[p] + ks * 32);
            aReg[p][0] = v.x; aReg[p][1] = v.y; aReg[p][2] = v.z; aReg[p][3] = v.w;
        }
        #pragma unroll
        for (int p = 0; p < 4; ++p) {
            const float4 v = *(const float4*)(W + (unsigned)((ks * 32 + kB + 8 * p) * ND + nB));
            bReg[p][0] = v.x; bReg[p][1] = v.y; bReg[p][2] = v.z; bReg[p][3] = v.w;
        }
    };

    auto storeLds = [&](int buf) {
        #pragma unroll
        for (int p = 0; p < 4; ++p) {
            const float sc = scv[p];
            uint2 pk;
            pk.x = pk2bf(aReg[p][0] * sc, aReg[p][1] * sc);
            pk.y = pk2bf(aReg[p][2] * sc, aReg[p][3] * sc);
            *(uint2*)&lA[buf][(rA + 32 * p) * LDS_STR + cA] = pk;   // ds_store_b64
        }
        #pragma unroll
        for (int p = 0; p < 4; ++p) {
            #pragma unroll
            for (int j = 0; j < 4; ++j)
                lB[buf][(nB + j) * LDS_STR + kB + 8 * p] = f2bf(bReg[p][j]);
        }
    };

    v8f acc[8];
    #pragma unroll
    for (int i = 0; i < 8; ++i)
        #pragma unroll
        for (int j = 0; j < 8; ++j) acc[i][j] = 0.0f;

    loadGlobal(0);
    storeLds(0);
    __syncthreads();

    const int arow = wv * 16 + l16;
    const int ka   = hi16 ? 8 : 0;
    const int kb   = hi16 ? 8 : 0;

    for (int ks = 0; ks < 4; ++ks) {
        if (ks < 3) loadGlobal(ks + 1);       // issue global loads early

        const int buf = ks & 1;
        us8 alo = *(const us8*)&lA[buf][arow * LDS_STR + ka];
        us8 ahi = *(const us8*)&lA[buf][arow * LDS_STR + ka + 16];
        v16bf afrag = mkfrag(alo, ahi);

        #pragma unroll
        for (int nt = 0; nt < 8; ++nt) {
            const int bcol = nt * 16 + l16;
            us8 blo = *(const us8*)&lB[buf][bcol * LDS_STR + kb];
            us8 bhi = *(const us8*)&lB[buf][bcol * LDS_STR + kb + 16];
            v16bf bfrag = mkfrag(blo, bhi);
            acc[nt] = __builtin_amdgcn_wmma_f32_16x16x32_bf16(
                false, afrag, false, bfrag, (short)0, acc[nt], false, false);
        }

        if (ks < 3) {
            storeLds((ks + 1) & 1);           // waits on loads; other buffer
            __syncthreads();
        }
    }

    // ---- epilogue ----
    float alpha = 0.f, onema = 0.f;
    if (MODE == 2) {
        const float s = skipPtr[0];
        alpha = 1.0f / (1.0f + __expf(-s));
        onema = 1.0f - alpha;
    }
    #pragma unroll
    for (int nt = 0; nt < 8; ++nt) {
        const int gc = nt * 16 + l16;
        const float b = bias[gc];
        #pragma unroll
        for (int j = 0; j < 8; ++j) {
            const int gr = rowBase + wv * 16 + j + hi16 * 8;
            if (gr < nrows) {
                const unsigned off = (unsigned)gr * ND + gc;
                float v = acc[nt][j] + b;
                if (MODE == 1)      v = tanhf(v);
                else if (MODE == 2) v = v * alpha + Xprev[off] * onema;
                Out[off] = v;
            }
        }
    }
}

// ---------------------------------------------------------------------------
// Per-relation init:  smax = -inf, ssum = 0   over N*H
// ---------------------------------------------------------------------------
__global__ __launch_bounds__(256)
void rel_init_kernel(float* __restrict__ smax, float* __restrict__ ssum)
{
    int i = blockIdx.x * 256 + threadIdx.x;
    if (i < NN * NH) { smax[i] = -1e30f; ssum[i] = 0.0f; }
}

// ---------------------------------------------------------------------------
// Edge attention score: att[e,h] = pri[h]/sqrt(32) * q_dst . (k_src @ ratt[h])
// Thread per (edge, head).  ratt (all heads) staged in LDS.
// ---------------------------------------------------------------------------
__global__ __launch_bounds__(256)
void edge_score_kernel(const float* __restrict__ Kb, const float* __restrict__ Qb,
                       const int* __restrict__ src, const int* __restrict__ dst,
                       const float* __restrict__ ratt, const float* __restrict__ pri,
                       float* __restrict__ attOut, float* __restrict__ smax)
{
    __shared__ float lR[NH * NDK * NDK];   // 16 KB
    for (int i = threadIdx.x; i < NH * NDK * NDK; i += 256) lR[i] = ratt[i];
    __syncthreads();

    const int idx = blockIdx.x * 256 + threadIdx.x;
    if (idx >= NE * NH) return;
    const int e = idx >> 2, h = idx & 3;
    const unsigned koff = (unsigned)src[e] * ND + h * NDK;
    const unsigned qoff = (unsigned)dst[e] * ND + h * NDK;
    const float* Rr = &lR[h * NDK * NDK];

    float kef[NDK];
    #pragma unroll
    for (int f = 0; f < NDK; ++f) kef[f] = 0.0f;
    for (int dd = 0; dd < NDK; ++dd) {
        const float kd = Kb[koff + dd];
        #pragma unroll
        for (int f4 = 0; f4 < NDK; f4 += 4) {
            const float4 rv = *(const float4*)&Rr[dd * NDK + f4];
            kef[f4 + 0] = fmaf(kd, rv.x, kef[f4 + 0]);
            kef[f4 + 1] = fmaf(kd, rv.y, kef[f4 + 1]);
            kef[f4 + 2] = fmaf(kd, rv.z, kef[f4 + 2]);
            kef[f4 + 3] = fmaf(kd, rv.w, kef[f4 + 3]);
        }
    }
    float att = 0.0f;
    #pragma unroll
    for (int f = 0; f < NDK; ++f) att = fmaf(Qb[qoff + f], kef[f], att);
    att = att * pri[h] * 0.17677669529663687f;   // 1/sqrt(32)

    attOut[idx] = att;
    atomicMaxF32(&smax[(unsigned)dst[e] * NH + h], att);
}

// ---------------------------------------------------------------------------
// exp(att - max), accumulate segment sums + per-dst degree.
// ---------------------------------------------------------------------------
__global__ __launch_bounds__(256)
void edge_expsum_kernel(const int* __restrict__ dst, float* __restrict__ att,
                        const float* __restrict__ smax, float* __restrict__ ssum,
                        float* __restrict__ deg)
{
    const int idx = blockIdx.x * 256 + threadIdx.x;
    if (idx >= NE * NH) return;
    const int e = idx >> 2, h = idx & 3;
    const unsigned d = (unsigned)dst[e];
    const float a = __expf(att[idx] - smax[d * NH + h]);
    att[idx] = a;
    atomicAdd(ssum + (d * NH + h), a);
    if (h == 0) atomicAdd(deg + d, 1.0f);
}

// ---------------------------------------------------------------------------
// Message pass via WMMA: per (16-edge chunk, head) a 16x32 x 32x32 bf16 GEMM
// (two v_wmma_f32_16x16x32_bf16), scaled by attn, atomically scattered to t_sum.
// Wave per group; 8 waves / block.
// ---------------------------------------------------------------------------
__global__ __launch_bounds__(256)
void edge_message_kernel(const float* __restrict__ Vb, const int* __restrict__ src,
                         const int* __restrict__ dst, const float* __restrict__ rmsg,
                         const float* __restrict__ att, const float* __restrict__ ssum,
                         float* __restrict__ TS, int ngroups)
{
    __shared__ __align__(16) unsigned short lRB[NH * 32 * LDS_STR]; // Bt per head
    __shared__ __align__(16) unsigned short lVA[8 * 16 * LDS_STR];  // A tiles per wave
    __shared__ float lAttn[8 * 16];
    __shared__ unsigned lDOf[8 * 16];   // precomputed dst row offsets (elements)

    const int tid  = threadIdx.x;
    const int wv   = tid >> 5;
    const int lane = tid & 31;
    const int hi16 = lane >> 4;
    const int l16  = lane & 15;

    // stage rmsg (all heads) transposed: lRB[h][f][d]
    for (int i = tid; i < NH * NDK * NDK; i += 256) {
        const int hh = i >> 10, rem = i & 1023, dd = rem >> 5, ff = rem & 31;
        lRB[hh * (32 * LDS_STR) + ff * LDS_STR + dd] = f2bf(rmsg[i]);
    }

    const int g = blockIdx.x * 8 + wv;
    const bool active = (g < ngroups);
    int h = 0, e0 = 0;
    if (active) {
        h  = g & 3;
        e0 = (g >> 2) * 16;
        // gather 16 V rows (head slice) -> bf16 LDS; lane covers half a row
        const int m = lane >> 1, part = lane & 1;
        const int e = e0 + m;
        const unsigned voff = (unsigned)src[e] * ND + h * NDK + part * 16;
        unsigned short* dp = &lVA[(wv * 16 + m) * LDS_STR + part * 16];
        #pragma unroll
        for (int i = 0; i < 2; ++i) {
            const float4 x0 = *(const float4*)(Vb + voff + i * 8);
            const float4 x1 = *(const float4*)(Vb + voff + i * 8 + 4);
            uint4 pk;
            pk.x = pk2bf(x0.x, x0.y); pk.y = pk2bf(x0.z, x0.w);
            pk.z = pk2bf(x1.x, x1.y); pk.w = pk2bf(x1.z, x1.w);
            *(uint4*)(dp + i * 8) = pk;                       // ds_store_b128
        }
        if (lane < 16) {
            const int e2 = e0 + lane;
            const unsigned dd = (unsigned)dst[e2];
            const float a = att[(unsigned)e2 * NH + h];
            lAttn[wv * 16 + lane] = a / ssum[dd * NH + h];
            lDOf [wv * 16 + lane] = dd * ND + h * NDK;        // 32-bit scatter base
        }
    }
    __syncthreads();
    if (!active) return;

    // A fragment (edges x k)
    const int arow = wv * 16 + l16;
    const int ka   = hi16 ? 8 : 0;
    us8 alo = *(const us8*)&lVA[arow * LDS_STR + ka];
    us8 ahi = *(const us8*)&lVA[arow * LDS_STR + ka + 16];
    v16bf afrag = mkfrag(alo, ahi);

    v8f acc[2];
    #pragma unroll
    for (int i = 0; i < 2; ++i)
        #pragma unroll
        for (int j = 0; j < 8; ++j) acc[i][j] = 0.0f;

    const unsigned short* bbase = &lRB[h * (32 * LDS_STR)];
    #pragma unroll
    for (int nt = 0; nt < 2; ++nt) {
        const int bcol = nt * 16 + l16;
        const int kb   = hi16 ? 8 : 0;
        us8 blo = *(const us8*)&bbase[bcol * LDS_STR + kb];
        us8 bhi = *(const us8*)&bbase[bcol * LDS_STR + kb + 16];
        v16bf bfrag = mkfrag(blo, bhi);
        acc[nt] = __builtin_amdgcn_wmma_f32_16x16x32_bf16(
            false, afrag, false, bfrag, (short)0, acc[nt], false, false);
    }

    // attn-scaled scatter-add into t_sum[dst]; 32-bit offsets vs uniform base
    #pragma unroll
    for (int nt = 0; nt < 2; ++nt) {
        const int fcol = nt * 16 + l16;
        #pragma unroll
        for (int j = 0; j < 8; ++j) {
            const int m  = j + hi16 * 8;
            const float w = lAttn[wv * 16 + m];
            const unsigned off = lDOf[wv * 16 + m] + fcol;
            atomicAdd(TS + off, acc[nt][j] * w);
        }
    }
}

// ---------------------------------------------------------------------------
// After one relation: cnt += (deg > 0); reset deg.
// ---------------------------------------------------------------------------
__global__ __launch_bounds__(256)
void rel_finalize_kernel(float* __restrict__ cnt, float* __restrict__ deg)
{
    int i = blockIdx.x * 256 + threadIdx.x;
    if (i < NN) {
        if (deg[i] > 0.5f) cnt[i] += 1.0f;
        deg[i] = 0.0f;
    }
}

__global__ __launch_bounds__(256)
void zero_kernel(float* __restrict__ p, long n)
{
    long i = (long)blockIdx.x * 256 + threadIdx.x;
    const long stride = (long)gridDim.x * 256;
    for (; i < n; i += stride) p[i] = 0.0f;
}

// ---------------------------------------------------------------------------
// LayerNorm: wave per row (4 floats / lane), __shfl_xor reductions.
// ---------------------------------------------------------------------------
__global__ __launch_bounds__(256)
void layernorm_kernel(const float* __restrict__ X, const float* __restrict__ g,
                      const float* __restrict__ b, float* __restrict__ Out,
                      int nrowsTotal)
{
    const int wv = threadIdx.x >> 5, lane = threadIdx.x & 31;
    const int row = blockIdx.x * 8 + wv;
    if (row >= nrowsTotal) return;
    const int t = row / NN;
    const unsigned roff = (unsigned)row * ND + lane * 4;

    const float4 v = *(const float4*)(X + roff);
    float s = v.x + v.y + v.z + v.w;
    #pragma unroll
    for (int o = 16; o >= 1; o >>= 1) s += __shfl_xor(s, o, 32);
    const float mu = s * (1.0f / 128.0f);
    const float dx = v.x - mu, dy = v.y - mu, dz = v.z - mu, dw = v.w - mu;
    float q = dx * dx + dy * dy + dz * dz + dw * dw;
    #pragma unroll
    for (int o = 16; o >= 1; o >>= 1) q += __shfl_xor(q, o, 32);
    const float rstd = rsqrtf(q * (1.0f / 128.0f) + 1e-5f);

    const unsigned goff = (unsigned)t * ND + lane * 4;
    const float4 gg = *(const float4*)(g + goff);
    const float4 bb = *(const float4*)(b + goff);
    float4 o;
    o.x = gg.x * dx * rstd + bb.x;
    o.y = gg.y * dy * rstd + bb.y;
    o.z = gg.z * dz * rstd + bb.z;
    o.w = gg.w * dw * rstd + bb.w;
    *(float4*)(Out + roff) = o;
}

// ---------------------------------------------------------------------------
// Host-side orchestration (graph-capture safe: only kernel launches on stream).
// ---------------------------------------------------------------------------
static const int kRelSrc[NR] = {0, 1, 2, 0, 1, 2};
static const int kRelDst[NR] = {1, 2, 0, 2, 0, 1};

extern "C" void kernel_launch(void* const* d_in, const int* in_sizes, int n_in,
                              void* d_out, int out_size, void* d_ws, size_t ws_size,
                              hipStream_t stream)
{
    (void)in_sizes; (void)n_in; (void)out_size; (void)ws_size;

    const int*   names    = (const int*)  d_in[0];
    const int*   src_idx  = (const int*)  d_in[1];
    const int*   dst_idx  = (const int*)  d_in[2];
    const float* node_emb = (const float*)d_in[3];
    const float* adapt_w  = (const float*)d_in[4];
    const float* adapt_b  = (const float*)d_in[5];
    const float* k_w      = (const float*)d_in[6];
    const float* k_b      = (const float*)d_in[7];
    const float* q_w      = (const float*)d_in[8];
    const float* q_b      = (const float*)d_in[9];
    const float* v_w      = (const float*)d_in[10];
    const float* v_b      = (const float*)d_in[11];
    const float* a_w      = (const float*)d_in[12];
    const float* a_b      = (const float*)d_in[13];
    const float* rel_pri  = (const float*)d_in[14];
    const float* rel_att  = (const float*)d_in[15];
    const float* rel_msg  = (const float*)d_in[16];
    const float* skip     = (const float*)d_in[17];
    const float* ln_g     = (const float*)d_in[18];
    const float* ln_b     = (const float*)d_in[19];
    float* outp = (float*)d_out;
    float* ws   = (float*)d_ws;

    const long SZ_H = (long)TT * NN * ND;                 // 38 400 000 floats
    float* H0   = ws;
    float* H1   = ws + SZ_H;
    float* Kb   = ws + 2 * SZ_H;
    float* Qb   = ws + 3 * SZ_H;
    float* Vb   = ws + 4 * SZ_H;
    float* TS   = ws + 5 * SZ_H;
    float* CNT  = ws + 6 * SZ_H;                          // TT*NN
    float* DEG  = CNT + (long)TT * NN;                    // NN
    float* ATT  = DEG + NN;                               // NE*NH
    float* SMAX = ATT + (long)NE * NH;                    // NN*NH
    float* SSUM = SMAX + (long)NN * NH;                   // NN*NH
    float* MIX  = Kb;  // reuse K buffer for pre-LN output

    const dim3 blk(256);
    const int gemmBlocks = (NN + 127) / 128;
    const int ehBlocks   = (NE * NH + 255) / 256;
    const int ngroups    = (NE / 16) * NH;
    const int msgBlocks  = (ngroups + 7) / 8;

    // -- adapt: H0[t] = tanh(node_emb[names[t]] @ adapt_w[t] + adapt_b[t]) --
    for (int t = 0; t < TT; ++t)
        wmma_gemm_kernel<1, true, false><<<gemmBlocks, blk, 0, stream>>>(
            node_emb, names + (long)t * NN, nullptr,
            adapt_w + (long)t * ND * ND, adapt_b + (long)t * ND,
            nullptr, nullptr, H0 + (long)t * NN * ND, NN);

    for (int l = 0; l < NL; ++l) {
        float* hin   = (l == 0) ? H0 : H1;
        float* lnout = (l == NL - 1) ? outp : H1;
        const long lw = (long)l * TT * ND * ND;
        const long lb = (long)l * TT * ND;

        // K/Q/V projections (bias only)
        for (int t = 0; t < TT; ++t) {
            const float* x = hin + (long)t * NN * ND;
            wmma_gemm_kernel<0, false, false><<<gemmBlocks, blk, 0, stream>>>(
                x, nullptr, nullptr, k_w + lw + (long)t * ND * ND,
                k_b + lb + (long)t * ND, nullptr, nullptr,
                Kb + (long)t * NN * ND, NN);
            wmma_gemm_kernel<0, false, false><<<gemmBlocks, blk, 0, stream>>>(
                x, nullptr, nullptr, q_w + lw + (long)t * ND * ND,
                q_b + lb + (long)t * ND, nullptr, nullptr,
                Qb + (long)t * NN * ND, NN);
            wmma_gemm_kernel<0, false, false><<<gemmBlocks, blk, 0, stream>>>(
                x, nullptr, nullptr, v_w + lw + (long)t * ND * ND,
                v_b + lb + (long)t * ND, nullptr, nullptr,
                Vb + (long)t * NN * ND, NN);
        }

        // zero t_sum + cnt + deg (contiguous region)
        zero_kernel<<<4096, blk, 0, stream>>>(TS, SZ_H + (long)TT * NN + NN);

        for (int r = 0; r < NR; ++r) {
            const int st = kRelSrc[r], dt = kRelDst[r];
            const float* ra  = rel_att + ((long)(l * NR + r)) * NH * NDK * NDK;
            const float* rm  = rel_msg + ((long)(l * NR + r)) * NH * NDK * NDK;
            const float* pr  = rel_pri + (long)(l * NR + r) * NH;
            const int*   se  = src_idx + (long)r * NE;
            const int*   de  = dst_idx + (long)r * NE;

            rel_init_kernel<<<(NN * NH + 255) / 256, blk, 0, stream>>>(SMAX, SSUM);
            edge_score_kernel<<<ehBlocks, blk, 0, stream>>>(
                Kb + (long)st * NN * ND, Qb + (long)dt * NN * ND,
                se, de, ra, pr, ATT, SMAX);
            edge_expsum_kernel<<<ehBlocks, blk, 0, stream>>>(de, ATT, SMAX, SSUM, DEG);
            edge_message_kernel<<<msgBlocks, blk, 0, stream>>>(
                Vb + (long)st * NN * ND, se, de, rm, ATT, SSUM,
                TS + (long)dt * NN * ND, ngroups);
            rel_finalize_kernel<<<(NN + 255) / 256, blk, 0, stream>>>(
                CNT + (long)dt * NN, DEG);
        }

        // A-projection with fused mean (rowScale=cnt) + sigmoid-skip residual
        for (int t = 0; t < TT; ++t)
            wmma_gemm_kernel<2, false, true><<<gemmBlocks, blk, 0, stream>>>(
                TS + (long)t * NN * ND, nullptr, CNT + (long)t * NN,
                a_w + lw + (long)t * ND * ND, a_b + lb + (long)t * ND,
                hin + (long)t * NN * ND, skip + (long)l * TT + t,
                MIX + (long)t * NN * ND, NN);

        // per-type LayerNorm
        layernorm_kernel<<<(TT * NN + 7) / 8, blk, 0, stream>>>(
            MIX, ln_g + (long)l * TT * ND, ln_b + (long)l * TT * ND,
            lnout, TT * NN);
    }
}
